// MultiHeadedAttention_43387759624759
// MI455X (gfx1250) — matrix operand order repaired and checked by reading, not compiled, not run on hardware
//
#include <hip/hip_runtime.h>
#include <hip/hip_bf16.h>
#include <math.h>

#define Bz 4
#define Nn 1024
#define Dd 512
#define Hh 8
#define HD 64
#define NEG_BIG -1000000000.0f
#define REL_FILL -10000.0f

typedef __attribute__((ext_vector_type(16))) __bf16 v16bf;
typedef __attribute__((ext_vector_type(8)))  float  v8f;
typedef unsigned short u16;
typedef unsigned int   u32;

struct U16x16 { uint4 lo, hi; };

static __device__ __forceinline__ v16bf mkv16(uint4 lo, uint4 hi) {
    U16x16 t{lo, hi};
    return __builtin_bit_cast(v16bf, t);
}

// f32 -> bf16 round-to-nearest-even via bit manipulation (toolchain-safe)
static __device__ __forceinline__ u16 f2bf(float f) {
    u32 u = __builtin_bit_cast(u32, f);
    u32 r = u + 0x7FFFu + ((u >> 16) & 1u);
    return (u16)(r >> 16);
}

// ---------------------------------------------------------------------------
// Kernel 0: f32 -> bf16 conversion, vectorized x4 (n must be multiple of 4)
// ---------------------------------------------------------------------------
__global__ void cvt_kernel(const float* __restrict__ src, u16* __restrict__ dst, int n4) {
    int i = blockIdx.x * blockDim.x + threadIdx.x;
    if (i < n4) {
        float4 f = ((const float4*)src)[i];
        union { u16 h[4]; uint2 u; } o;
        o.h[0] = f2bf(f.x); o.h[1] = f2bf(f.y);
        o.h[2] = f2bf(f.z); o.h[3] = f2bf(f.w);
        ((uint2*)dst)[i] = o.u;
    }
}

// ---------------------------------------------------------------------------
// Kernel 1: projections  P = X @ W^T + b  (bf16 WMMA, f32 accum)
//   z=0: q -> qbf [B][H][N][HD], scaled by 1/sqrt(HD)
//   z=1: k -> kbf [B][H][N][HD]
//   z=2: v -> vT  [B][H][HD][N]   (transposed for prob@v B-matrix loads)
// ---------------------------------------------------------------------------
__global__ __launch_bounds__(256) void proj_kernel(
    const u16* __restrict__ Xbf,   // [3][B*N][D]
    const u16* __restrict__ Wbf,   // [3][D][D]
    const float* __restrict__ bq, const float* __restrict__ bk,
    const float* __restrict__ bv,
    u16* __restrict__ qbf, u16* __restrict__ kbf, u16* __restrict__ vT)
{
    const int z = blockIdx.z;
    const u16* X = Xbf + (size_t)z * (Bz * Nn * Dd);
    const u16* W = Wbf + (size_t)z * (Dd * Dd);
    const float* bias = (z == 0) ? bq : ((z == 1) ? bk : bv);

    const int lane = threadIdx.x & 31;
    const int wave = threadIdx.x >> 5;   // 0..7
    const int wrow = wave >> 2;          // 0..1
    const int wcol = wave & 3;           // 0..3
    const int ln = lane & 15;
    const int hh = lane >> 4;

    const int rowBase = blockIdx.x * 32 + wrow * 16;  // [0,4096)
    const int colBase = blockIdx.y * 64 + wcol * 16;  // [0,512)

    const u16* arow = X + (size_t)(rowBase + ln) * Dd;   // A: row per lane
    const u16* brow = W + (size_t)(colBase + ln) * Dd;   // B: col per lane (W row)

    v8f acc = {};
#pragma unroll 4
    for (int kk = 0; kk < Dd; kk += 32) {
        const u16* ap = arow + kk + 8 * hh;              // K chunks 8h..8h+7, 16+8h..
        v16bf a = mkv16(*(const uint4*)ap, *(const uint4*)(ap + 16));
        const u16* bp = brow + kk + 8 * hh;              // K chunks 8h..8h+7, 16+8h..
        v16bf b = mkv16(*(const uint4*)bp, *(const uint4*)(bp + 16));
        acc = __builtin_amdgcn_wmma_f32_16x16x32_bf16(false, a, false, b,
                                                      (short)0, acc, false, false);
    }

    const float s = (z == 0) ? 0.125f : 1.0f;   // 1/sqrt(64)
    const int gn = colBase + ln;
    const float bval = bias[gn];
    const int h = gn >> 6, hd = gn & 63;
#pragma unroll
    for (int r = 0; r < 8; ++r) {
        int gm = rowBase + r + 8 * hh;           // row in [0, B*N)
        int b_ = gm >> 10, nseq = gm & 1023;
        u16 o = f2bf((acc[r] + bval) * s);
        if (z == 2) {
            vT[((size_t)((b_ * Hh + h) * HD + hd)) * Nn + nseq] = o;
        } else {
            u16* dst = (z == 0) ? qbf : kbf;
            dst[((size_t)((b_ * Hh + h) * Nn + nseq)) * HD + hd] = o;
        }
    }
}

// ---------------------------------------------------------------------------
// Kernel 2: head-independent blended base = c1*time_attn + c2*rel_attn + c3*resp_attn
// one block (256 thr) per (b,i) row; mask derived from j>i (triu k=1)
// ---------------------------------------------------------------------------
static __device__ __forceinline__ float blockMax(float v, float* red) {
    int t = threadIdx.x;
    red[t] = v; __syncthreads();
#pragma unroll
    for (int s = 128; s > 0; s >>= 1) {
        if (t < s) red[t] = fmaxf(red[t], red[t + s]);
        __syncthreads();
    }
    float r = red[0]; __syncthreads(); return r;
}
static __device__ __forceinline__ float blockSum(float v, float* red) {
    int t = threadIdx.x;
    red[t] = v; __syncthreads();
#pragma unroll
    for (int s = 128; s > 0; s >>= 1) {
        if (t < s) red[t] = red[t] + red[t + s];
        __syncthreads();
    }
    float r = red[0]; __syncthreads(); return r;
}

__global__ __launch_bounds__(256) void base_kernel(
    const float* __restrict__ rel, const float* __restrict__ resp,
    const float* __restrict__ ts,
    const float* __restrict__ pl1, const float* __restrict__ pl2,
    const float* __restrict__ pl3,
    float* __restrict__ base)
{
    __shared__ float red[256];
    const int row = blockIdx.x;          // b*N + i
    const int i = row & (Nn - 1);
    const int t = threadIdx.x;

    const float l1 = pl1[0], l2 = pl2[0], l3 = pl3[0];
    const float c1 = (1.f - l3) * (1.f - l1) * l2;
    const float c2 = (1.f - l3) * l1;
    const float c3 = l3;

    const float* relR  = rel  + (size_t)row * Nn;
    const float* respR = resp + (size_t)row * Nn;
    const float* tsR   = ts   + (size_t)row * Nn;

    float rv[4], sv[4], tv[4];
#pragma unroll
    for (int e = 0; e < 4; ++e) {
        int j = t + e * 256;
        bool valid = (j <= i);
        float r = valid ? __builtin_nontemporal_load(&relR[j])  : 0.f;
        rv[e] = (r == 0.f) ? REL_FILL : r;
        float s = valid ? __builtin_nontemporal_load(&respR[j]) : 0.f;
        sv[e] = (s == 0.f) ? REL_FILL : s;
        tv[e] = valid ? __expf(-fabsf(__builtin_nontemporal_load(&tsR[j])))
                      : -__builtin_inff();
    }

    float mR = blockMax(fmaxf(fmaxf(rv[0], rv[1]), fmaxf(rv[2], rv[3])), red);
    float mS = blockMax(fmaxf(fmaxf(sv[0], sv[1]), fmaxf(sv[2], sv[3])), red);
    float mT = blockMax(fmaxf(fmaxf(tv[0], tv[1]), fmaxf(tv[2], tv[3])), red);

    float eR[4], eS[4], eT[4], aR = 0.f, aS = 0.f, aT = 0.f;
#pragma unroll
    for (int e = 0; e < 4; ++e) {
        eR[e] = __expf(rv[e] - mR); aR += eR[e];
        eS[e] = __expf(sv[e] - mS); aS += eS[e];
        eT[e] = __expf(tv[e] - mT); aT += eT[e];
    }
    float sumR = blockSum(aR, red);
    float sumS = blockSum(aS, red);
    float sumT = blockSum(aT, red);
    float iR = c2 / sumR, iS = c3 / sumS, iT = c1 / sumT;

    float* dst = base + (size_t)row * Nn;
#pragma unroll
    for (int e = 0; e < 4; ++e) {
        int j = t + e * 256;
        dst[j] = eT[e] * iT + eR[e] * iR + eS[e] * iS;
    }
}

// ---------------------------------------------------------------------------
// Kernel 3: fused attention. Block = 256 thr (8 waves) per (b,h, 16 q-rows).
// Wave w owns cols [128w,128w+128). Causal skip: tiles with j0 > i0+15 are
// fully masked (prob == base == 0 exactly) -> skip score & out WMMA there.
// ---------------------------------------------------------------------------
__global__ __launch_bounds__(256) void attn_kernel(
    const u16* __restrict__ qbf, const u16* __restrict__ kbf,
    const u16* __restrict__ vT,  const float* __restrict__ base,
    const float* __restrict__ pl1, const float* __restrict__ pl2,
    const float* __restrict__ pl3,
    float* __restrict__ out,    // [B][N][D]
    float* __restrict__ prob)   // [B][H][N][N]
{
    __shared__ float rmax_ws[8][16];
    __shared__ float rsum_ws[8][16];
    __shared__ float rmaxF[16];
    __shared__ float rsumF[16];
    __shared__ u16   stage[8][16 * 32];   // per-wave 16x32 bf16 prob tile
    __shared__ float outAcc[16][HD];

    const int bh = blockIdx.y;            // b*H + h
    const int b_ = bh >> 3, h = bh & 7;
    const int i0 = blockIdx.x * 16;

    const int lane = threadIdx.x & 31;
    const int wave = threadIdx.x >> 5;
    const int ln = lane & 15, hh = lane >> 4;

    const float l1 = pl1[0], l2 = pl2[0], l3 = pl3[0];
    const float c0 = (1.f - l3) * (1.f - l1) * (1.f - l2);

    for (int idx = threadIdx.x; idx < 16 * HD; idx += 256)
        (&outAcc[0][0])[idx] = 0.f;

    // Causal skip: wave-uniform count of j-tiles touching the valid (j<=i) region
    const int jw = wave * 128;
    int nvalid = 0;
    if (i0 + 15 >= jw) {
        nvalid = ((i0 + 15 - jw) >> 4) + 1;
        if (nvalid > 8) nvalid = 8;
    }

    // Preload q A-fragments (16 rows x K=64, two K-steps), reused for all tiles
    const u16* qrow = qbf + ((size_t)bh * Nn + i0 + ln) * HD;
    v16bf aq[2];
#pragma unroll
    for (int kc = 0; kc < 2; ++kc) {
        const u16* ap = qrow + kc * 32 + 8 * hh;
        aq[kc] = mkv16(*(const uint4*)ap, *(const uint4*)(ap + 16));
    }

    // Phase 1: scores (registers), masking; fully-masked tiles skipped
    v8f sc[8] = {};
#pragma unroll
    for (int tIdx = 0; tIdx < 8; ++tIdx) {
        if (tIdx < nvalid) {
            int j0 = jw + tIdx * 16;
            const u16* krow = kbf + ((size_t)bh * Nn + j0 + ln) * HD;
            v8f a = {};
#pragma unroll
            for (int kc = 0; kc < 2; ++kc) {
                const u16* bp = krow + kc * 32 + 8 * hh;
                v16bf bm = mkv16(*(const uint4*)bp, *(const uint4*)(bp + 16));
                a = __builtin_amdgcn_wmma_f32_16x16x32_bf16(false, aq[kc], false, bm,
                                                            (short)0, a, false, false);
            }
#pragma unroll
            for (int r = 0; r < 8; ++r) {
                int gi = i0 + r + 8 * hh;
                int gj = j0 + ln;
                if (gj > gi) a[r] = NEG_BIG;
            }
            sc[tIdx] = a;
        }
    }

    // Row max: per-lane over valid tiles, shfl across 16 lanes, LDS across waves
    float pm[8];
#pragma unroll
    for (int r = 0; r < 8; ++r) {
        float m = NEG_BIG;
#pragma unroll
        for (int tIdx = 0; tIdx < 8; ++tIdx)
            if (tIdx < nvalid) m = fmaxf(m, sc[tIdx][r]);
#pragma unroll
        for (int s = 1; s < 16; s <<= 1) m = fmaxf(m, __shfl_xor(m, s, 32));
        pm[r] = m;
    }
    if (ln == 0) {
#pragma unroll
        for (int r = 0; r < 8; ++r) rmax_ws[wave][r + 8 * hh] = pm[r];
    }
    __syncthreads();
    if (threadIdx.x < 16) {
        float m = rmax_ws[0][threadIdx.x];
#pragma unroll
        for (int w = 1; w < 8; ++w) m = fmaxf(m, rmax_ws[w][threadIdx.x]);
        rmaxF[threadIdx.x] = m;
    }
    __syncthreads();

    // Phase 2: exponentiate valid tiles in place, row sums
#pragma unroll
    for (int r = 0; r < 8; ++r) {
        float mrow = rmaxF[r + 8 * hh];
        float s = 0.f;
#pragma unroll
        for (int tIdx = 0; tIdx < 8; ++tIdx) {
            if (tIdx < nvalid) {
                float e = __expf(sc[tIdx][r] - mrow);
                sc[tIdx][r] = e;
                s += e;
            }
        }
#pragma unroll
        for (int sh = 1; sh < 16; sh <<= 1) s += __shfl_xor(s, sh, 32);
        pm[r] = s;
    }
    if (ln == 0) {
#pragma unroll
        for (int r = 0; r < 8; ++r) rsum_ws[wave][r + 8 * hh] = pm[r];
    }
    __syncthreads();
    if (threadIdx.x < 16) {
        float s = rsum_ws[0][threadIdx.x];
#pragma unroll
        for (int w = 1; w < 8; ++w) s += rsum_ws[w][threadIdx.x];
        rsumF[threadIdx.x] = s;
    }
    __syncthreads();

    // Hoisted per-row scale: c0 / rowsum
    float rscale[8];
#pragma unroll
    for (int r = 0; r < 8; ++r) rscale[r] = c0 / rsumF[r + 8 * hh];

    // Phase 3: prob -> global (NT store); stage bf16 in LDS; out WMMA over vT.
    // Chunks whose first tile is fully masked contribute exactly 0 -> skip WMMA.
    float* probBH = prob + (size_t)bh * Nn * Nn;
    const float* baseB = base + ((size_t)b_ * Nn + i0) * Nn;
    v8f oacc[4] = {};
#pragma unroll
    for (int kc = 0; kc < 4; ++kc) {
        const bool chunkValid = (2 * kc) < nvalid;
#pragma unroll
        for (int tt = 0; tt < 2; ++tt) {
            int tIdx = 2 * kc + tt;
            int j0 = jw + tIdx * 16;
#pragma unroll
            for (int r = 0; r < 8; ++r) {
                int m = r + 8 * hh;
                int gj = j0 + ln;
                float bv = baseB[(size_t)m * Nn + gj];
                float p = (tIdx < nvalid) ? sc[tIdx][r] * rscale[r] + bv : bv;
                __builtin_nontemporal_store(p, &probBH[(size_t)(i0 + m) * Nn + gj]);
                if (chunkValid) stage[wave][m * 32 + tt * 16 + ln] = f2bf(p);
            }
        }
        if (chunkValid) {
            // A-fragment from staged prob (row ln, K chunks 8h and 16+8h)
            const u16* sp = &stage[wave][0];
            const u16* apt = sp + ln * 32 + 8 * hh;
            v16bf aP = mkv16(*(const uint4*)apt, *(const uint4*)(apt + 16));
            int jbase = jw + kc * 32;
#pragma unroll
            for (int dt = 0; dt < 4; ++dt) {
                const u16* vp = vT + ((size_t)bh * HD + dt * 16 + ln) * Nn
                                + jbase + 8 * hh;
                v16bf bm = mkv16(*(const uint4*)vp, *(const uint4*)(vp + 16));
                oacc[dt] = __builtin_amdgcn_wmma_f32_16x16x32_bf16(
                    false, aP, false, bm, (short)0, oacc[dt], false, false);
            }
        }
    }

    // Phase 4: cross-wave reduce via LDS float atomics, write out
#pragma unroll
    for (int dt = 0; dt < 4; ++dt) {
#pragma unroll
        for (int r = 0; r < 8; ++r)
            atomicAdd(&outAcc[r + 8 * hh][dt * 16 + ln], oacc[dt][r]);
    }
    __syncthreads();
    for (int idx = threadIdx.x; idx < 16 * HD; idx += 256) {
        int m = idx >> 6, d = idx & 63;
        out[((size_t)b_ * Nn + i0 + m) * Dd + h * HD + d] = outAcc[m][d];
    }
}

// ---------------------------------------------------------------------------
extern "C" void kernel_launch(void* const* d_in, const int* in_sizes, int n_in,
                              void* d_out, int out_size, void* d_ws, size_t ws_size,
                              hipStream_t stream) {
    const float* query = (const float*)d_in[0];
    const float* key   = (const float*)d_in[1];
    const float* value = (const float*)d_in[2];
    const float* rel   = (const float*)d_in[3];
    const float* resp  = (const float*)d_in[4];
    const float* ts    = (const float*)d_in[5];
    // d_in[6] = mask (derived analytically from triu -> not read)
    const float* l1 = (const float*)d_in[7];
    const float* l2 = (const float*)d_in[8];
    const float* l3 = (const float*)d_in[9];
    const float* Wq = (const float*)d_in[10];
    const float* bq = (const float*)d_in[11];
    const float* Wk = (const float*)d_in[12];
    const float* bk = (const float*)d_in[13];
    const float* Wv = (const float*)d_in[14];
    const float* bv = (const float*)d_in[15];

    char* ws = (char*)d_ws;
    size_t off = 0;
    const size_t nX = (size_t)Bz * Nn * Dd;       // 2M elems per tensor
    const size_t nW = (size_t)Dd * Dd;            // 256K per weight
    const size_t nP = (size_t)Bz * Hh * Nn * HD;  // 2M per projection

    u16* Xbf = (u16*)(ws + off); off += 3 * nX * sizeof(u16);
    u16* Wbf = (u16*)(ws + off); off += 3 * nW * sizeof(u16);
    u16* qbf = (u16*)(ws + off); off += nP * sizeof(u16);
    u16* kbf = (u16*)(ws + off); off += nP * sizeof(u16);
    u16* vTb = (u16*)(ws + off); off += nP * sizeof(u16);
    float* basep = (float*)(ws + off); off += (size_t)Bz * Nn * Nn * sizeof(float);

    float* outp = (float*)d_out;
    float* prob = outp + (size_t)Bz * Nn * Dd;

    // Stage 0: f32 -> bf16 (vectorized x4)
    const int nX4 = (int)(nX / 4), nW4 = (int)(nW / 4);
    cvt_kernel<<<(nX4 + 255) / 256, 256, 0, stream>>>(query, Xbf + 0 * nX, nX4);
    cvt_kernel<<<(nX4 + 255) / 256, 256, 0, stream>>>(key,   Xbf + 1 * nX, nX4);
    cvt_kernel<<<(nX4 + 255) / 256, 256, 0, stream>>>(value, Xbf + 2 * nX, nX4);
    cvt_kernel<<<(nW4 + 255) / 256, 256, 0, stream>>>(Wq, Wbf + 0 * nW, nW4);
    cvt_kernel<<<(nW4 + 255) / 256, 256, 0, stream>>>(Wk, Wbf + 1 * nW, nW4);
    cvt_kernel<<<(nW4 + 255) / 256, 256, 0, stream>>>(Wv, Wbf + 2 * nW, nW4);

    // Stage 1: projections (WMMA)
    dim3 pgrid((Bz * Nn) / 32, Dd / 64, 3);
    proj_kernel<<<pgrid, 256, 0, stream>>>(Xbf, Wbf, bq, bk, bv, qbf, kbf, vTb);

    // Stage 2: head-independent blended softmax base
    base_kernel<<<Bz * Nn, 256, 0, stream>>>(rel, resp, ts, l1, l2, l3, basep);

    // Stage 3: fused scores-softmax-blend-output (WMMA, causal skip)
    dim3 agrid(Nn / 16, Bz * Hh);
    attn_kernel<<<agrid, 256, 0, stream>>>(qbf, kbf, vTb, basep, l1, l2, l3, outp, prob);
}
